// MultiHeadDifferentialAttention_10814727651698
// MI455X (gfx1250) — compile-verified, matrix operand order checked by reading, not run on hardware
//
#include <hip/hip_runtime.h>
#include <hip/hip_bf16.h>
#include <math.h>

typedef unsigned short us;
typedef unsigned int   u32;
typedef __attribute__((ext_vector_type(16))) __bf16 v16bf;
typedef __attribute__((ext_vector_type(8)))  float  v8f;
typedef __attribute__((ext_vector_type(4)))  unsigned int u32x4;
typedef __attribute__((ext_vector_type(8)))  int  i32x8;
typedef __attribute__((ext_vector_type(4)))  int  i32x4;

#define HEADS   8
#define DHEAD   128
#define DH2     256            // 2*DHEAD
#define BATCH   2
#define SEQ     2048
#define DMODEL  1024
#define DE      2048           // 2*DHEAD*HEADS
#define LAMBDA_INIT 0.35550906759096926f
#define RMS_EPS     1e-5f
#define SCALING     0.08838834764831845f   // 1/sqrt(128)

#if defined(__has_builtin)
#if __has_builtin(__builtin_amdgcn_tensor_load_to_lds)
#define HAVE_TDM 1
#endif
#endif

union Frag { v16bf v; u32 u[8]; };

__device__ __forceinline__ us f2bf(float f){
  u32 x = __float_as_uint(f);
  x += 0x7FFFu + ((x >> 16) & 1u);      // round-to-nearest-even
  return (us)(x >> 16);
}

__device__ __forceinline__ int kidx(int g, int half){
  // 16-bit A/B operand: VGPR g holds K pair base = (g<4?0:16) + half*8 + (g%4)*2
  return ((g & 4) << 2) + half * 8 + ((g & 3) << 1);
}

// Load a 16x32 bf16 operand fragment from row-major [16][stride] (stride even).
// Lane L -> row L%16; K pairs contiguous -> one b32 load each (b128-coalescible).
__device__ __forceinline__ Frag load_frag(const us* p, int stride){
  int lane = threadIdx.x & 31;
  const us* rp = p + (size_t)(lane & 15) * stride;
  int half = lane >> 4;
  Frag f;
#pragma unroll
  for (int g = 0; g < 8; ++g)
    f.u[g] = *(const u32*)(rp + kidx(g, half));
  return f;
}

__device__ __forceinline__ v8f wmma_bf16(Frag a, Frag b, v8f c){
  return __builtin_amdgcn_wmma_f32_16x16x32_bf16(false, a.v, false, b.v,
                                                 (short)0, c, false, false);
}

#ifdef HAVE_TDM
__device__ __forceinline__ u32 lds_off(const void* p){
  return (u32)(size_t)(__attribute__((address_space(3))) const char*)p;
}

// Issue a 2D TDM tile load: tile_d0 x tile_d1 elements of 2 bytes,
// global row stride = stride_d0 elements, dense into LDS at lds_addr.
__device__ __forceinline__ void tdm_load_2d(u32 lds_addr, const us* gptr,
                                            u32 tile_d0, u32 tile_d1, u32 stride_d0){
  unsigned long long ga = (unsigned long long)(size_t)gptr;
  u32x4 g0;
  g0[0] = 1u;                              // count=1 (valid user descriptor)
  g0[1] = lds_addr;                        // bits  63:32  lds_addr
  g0[2] = (u32)ga;                         // bits  95:64  global_addr[31:0]
  g0[3] = ((u32)(ga >> 32) & 0x01FFFFFFu)  // bits 120:96  global_addr[56:32]
          | (2u << 30);                    // bits 127:126 type=2 ("image")
  i32x8 g1;
  g1[0] = (int)(1u << 16);                 // data_size=1 -> 2 bytes; wg_mask=0
  g1[1] = (int)(tile_d0 << 16);            // tensor_dim0[15:0] in d1[31:16]
  g1[2] = (int)(tile_d1 << 16);            // tensor_dim0 hi=0; tensor_dim1 lo
  g1[3] = (int)(tile_d0 << 16);            // tensor_dim1 hi=0; tile_dim0
  g1[4] = (int)tile_d1;                    // tile_dim1; tile_dim2=0
  g1[5] = (int)stride_d0;                  // tensor_dim0_stride[31:0]
  g1[6] = 0;                               // stride hi; dim1_stride lo
  g1[7] = 0;
  i32x4 z4 = {0, 0, 0, 0};
#if __has_include(<hip/amd_detail/amd_gfx1250_TDM.h>)
  i32x8 z8 = {0, 0, 0, 0, 0, 0, 0, 0};
  __builtin_amdgcn_tensor_load_to_lds(g0, g1, z4, z4, z8, 0);
#else
  __builtin_amdgcn_tensor_load_to_lds(g0, g1, z4, z4, 0);
#endif
}
#endif // HAVE_TDM

// ---------------------------------------------------------------- conversions
__global__ __launch_bounds__(256) void cvt_bf16(const float* __restrict__ in,
                                                us* __restrict__ out, int n){
  int i = blockIdx.x * 256 + threadIdx.x;
  if (i < n) out[i] = f2bf(in[i]);
}

__global__ void lambda_kernel(const float* __restrict__ lq1, const float* __restrict__ lk1,
                              const float* __restrict__ lq2, const float* __restrict__ lk2,
                              float* __restrict__ lam){
  int h = threadIdx.x;
  if (h < HEADS){
    float s1 = 0.f, s2 = 0.f;
    for (int i = 0; i < DHEAD; ++i){
      s1 += lq1[h*DHEAD+i] * lk1[h*DHEAD+i];
      s2 += lq2[h*DHEAD+i] * lk2[h*DHEAD+i];
    }
    lam[h] = __expf(s1) - __expf(s2) + LAMBDA_INIT;
  }
}

// ------------------------------------------------- WMMA GEMM: C = A(MxK) * B(NxK)^T
// mode 0: scatter bf16 output into per-head (B,H,N,256) layout
// mode 1: plain f32 row-major output
__global__ __launch_bounds__(256) void gemm_bf16_tn(const us* __restrict__ A,
                                                    const us* __restrict__ Bm,
                                                    int K, int Ncols, int mode,
                                                    us* __restrict__ outb,
                                                    float* __restrict__ outf){
  __shared__ us As[128*32];
  __shared__ us Bs[128*32];
  int tid  = threadIdx.x;
  int lane = tid & 31, wave = tid >> 5;
  int wr = wave & 3;     // 4 wave-rows  * 32 rows
  int wc = wave >> 2;    // 2 wave-cols  * 64 cols
  int m0 = blockIdx.x * 128, n0 = blockIdx.y * 128;

  v8f zero = {};
  v8f acc[2][4];
#pragma unroll
  for (int i = 0; i < 2; ++i)
#pragma unroll
    for (int j = 0; j < 4; ++j) acc[i][j] = zero;

  int uc = tid & 15, r0 = tid >> 4;
  int Ku = K >> 1;
  for (int k0 = 0; k0 < K; k0 += 32){
    const u32* ga = (const u32*)(A  + (size_t)m0 * K + k0);
    const u32* gb = (const u32*)(Bm + (size_t)n0 * K + k0);
    u32* sa = (u32*)As;
    u32* sb = (u32*)Bs;
#pragma unroll
    for (int i = 0; i < 8; ++i){
      int r = r0 + i * 16;
      sa[r*16 + uc] = ga[(size_t)r * Ku + uc];
      sb[r*16 + uc] = gb[(size_t)r * Ku + uc];
    }
    __syncthreads();
    if (k0 + 32 < K){
      __builtin_prefetch(A  + (size_t)(m0 + r0*8) * K + k0 + 32, 0, 0);
      __builtin_prefetch(Bm + (size_t)(n0 + r0*8) * K + k0 + 32, 0, 0);
    }
    Frag af[2], bf[4];
#pragma unroll
    for (int i = 0; i < 2; ++i) af[i] = load_frag(As + (wr*32 + i*16)*32, 32);
#pragma unroll
    for (int j = 0; j < 4; ++j) bf[j] = load_frag(Bs + (wc*64 + j*16)*32, 32);
#pragma unroll
    for (int i = 0; i < 2; ++i)
#pragma unroll
      for (int j = 0; j < 4; ++j)
        acc[i][j] = wmma_bf16(af[i], bf[j], acc[i][j]);
    __syncthreads();
  }

#pragma unroll
  for (int i = 0; i < 2; ++i)
#pragma unroll
    for (int j = 0; j < 4; ++j)
#pragma unroll
      for (int g = 0; g < 8; ++g){
        int row = m0 + wr*32 + i*16 + (lane >> 4)*8 + g;   // C/D layout
        int col = n0 + wc*64 + j*16 + (lane & 15);
        float v = acc[i][j][g];
        if (mode == 0){
          int b = row >> 11, n = row & (SEQ - 1);          // row = b*SEQ + n
          int h = col >> 8,  c = col & (DH2 - 1);          // col = h*256 + c
          outb[(((size_t)(b*HEADS + h) * SEQ) + n) * DH2 + c] = f2bf(v);
        } else {
          outf[(size_t)row * Ncols + col] = v;
        }
      }
}

// ----------------------------------- flash-attention single-softmax pass
// O[bh, q, :256] = softmax( Q_sel K_sel^T / sqrt(dh) ) @ V       (f32 out)
// Block = 4 waves; each wave owns a 16-row q tile; K/V k-chunks of 32 rows
// are staged once per block into LDS (TDM when available).
__global__ __launch_bounds__(128) void diff_attn_pass(const us* __restrict__ Qb,
                                                      const us* __restrict__ Kb,
                                                      const us* __restrict__ Vb,
                                                      float* __restrict__ O,
                                                      int sel){
  __shared__ us Ks[32*128];      // K chunk, [k][d] row-major (sel half)
  __shared__ us Vt[256*32];      // V chunk transposed, [c][k]
#ifdef HAVE_TDM
  __shared__ us Vs[32*256];      // V chunk row-major TDM landing buffer
#endif
  __shared__ us Pt[4][16*32];    // per-wave P bounce tile

  int tid  = threadIdx.x;
  int lane = tid & 31, wv = tid >> 5;
  int bh = blockIdx.y;
  int q0 = (blockIdx.x * 4 + wv) * 16;

  const us* qbase = Qb + ((size_t)bh * SEQ + q0) * DH2 + sel * DHEAD;
  const us* kb0   = Kb +  (size_t)bh * SEQ * DH2 + sel * DHEAD;
  const us* vb0   = Vb +  (size_t)bh * SEQ * DH2;

  Frag qf[4];                    // 16 x 128 of Q, 4 d-steps of 32
#pragma unroll
  for (int t = 0; t < 4; ++t) qf[t] = load_frag(qbase + t*32, DH2);

  v8f zero = {};
  v8f o[16];                     // 16 x 256 f32 accumulator
#pragma unroll
  for (int t = 0; t < 16; ++t) o[t] = zero;
  float m[8], l[8];
#pragma unroll
  for (int g = 0; g < 8; ++g){ m[g] = -3.0e38f; l[g] = 0.f; }

  for (int kt = 0; kt < SEQ; kt += 32){
    const us* gk = kb0 + (size_t)kt * DH2;
    const us* gv = vb0 + (size_t)kt * DH2;
    __syncthreads();                         // previous chunk fully consumed
#ifdef HAVE_TDM
    if (wv == 0){
      tdm_load_2d(lds_off(Ks), gk, 128, 32, DH2);   // 32x128 tile
      tdm_load_2d(lds_off(Vs), gv, 256, 32, DH2);   // 32x256 tile
      __builtin_amdgcn_s_wait_tensorcnt(0);
    }
    __syncthreads();                         // publish TDM data to all waves
    // transpose Vs[32][256] -> Vt[256][32] (LDS->LDS, DS only)
#pragma unroll
    for (int i = 0; i < 32; ++i){
      int idx = tid + i*128;                 // 0..4095 u32 words
      int k = idx >> 7, c2 = idx & 127;
      u32 w = ((const u32*)Vs)[idx];
      Vt[(c2*2)  *32 + k] = (us)(w & 0xffffu);
      Vt[(c2*2+1)*32 + k] = (us)(w >> 16);
    }
#else
    // manual coalesced staging
#pragma unroll
    for (int i = 0; i < 16; ++i){
      int idx = tid + i*128;                 // 0..2047 u32 words of K
      int r = idx >> 6, c = idx & 63;
      ((u32*)Ks)[idx] = *(const u32*)(gk + (size_t)r*DH2 + c*2);
    }
#pragma unroll
    for (int i = 0; i < 32; ++i){
      int idx = tid + i*128;                 // 0..4095 u32 words of V
      int k = idx >> 7, c2 = idx & 127;
      u32 w = *(const u32*)(gv + (size_t)k*DH2 + c2*2);
      Vt[(c2*2)  *32 + k] = (us)(w & 0xffffu);
      Vt[(c2*2+1)*32 + k] = (us)(w >> 16);
    }
#endif
    __syncthreads();                         // staged tiles ready

    // S = Q K^T for 16q x 32k (two 16x16 tiles), contraction over dh=128
    v8f s0 = zero, s1 = zero;
#pragma unroll
    for (int t = 0; t < 4; ++t){
      Frag b0 = load_frag(Ks + 0*128  + t*32, 128);
      Frag b1 = load_frag(Ks + 16*128 + t*32, 128);
      s0 = wmma_bf16(qf[t], b0, s0);
      s1 = wmma_bf16(qf[t], b1, s1);
    }
    // online softmax; rows live at (g, half) -> reduce across 16-lane halves
    float cr[8];
#pragma unroll
    for (int g = 0; g < 8; ++g){
      float a0 = s0[g] * SCALING;
      float a1 = s1[g] * SCALING;
      float mx = fmaxf(a0, a1);
#pragma unroll
      for (int off = 1; off < 16; off <<= 1) mx = fmaxf(mx, __shfl_xor(mx, off, 32));
      float mn = fmaxf(m[g], mx);
      float c  = __expf(m[g] - mn);
      float p0 = __expf(a0 - mn);
      float p1 = __expf(a1 - mn);
      float ps = p0 + p1;
#pragma unroll
      for (int off = 1; off < 16; off <<= 1) ps += __shfl_xor(ps, off, 32);
      l[g] = l[g] * c + ps;
      m[g] = mn;
      cr[g] = c;
      int row = (lane >> 4) * 8 + g;         // C-layout -> row-major P in LDS
      int col = lane & 15;
      Pt[wv][row*32 + col]      = f2bf(p0);
      Pt[wv][row*32 + col + 16] = f2bf(p1);
    }
#pragma unroll
    for (int t = 0; t < 16; ++t)
#pragma unroll
      for (int g = 0; g < 8; ++g) o[t][g] = o[t][g] * cr[g];

    // per-wave LDS store->load; same-wave DS ops are in order, wait DScnt
    asm volatile("s_wait_dscnt 0" ::: "memory");
    Frag pf = load_frag(&Pt[wv][0], 32);     // P as 16x32 A operand

#pragma unroll
    for (int t = 0; t < 16; ++t){
      Frag vf = load_frag(Vt + (t*16)*32, 32);  // V^T chunk as 32x16 B operand
      o[t] = wmma_bf16(pf, vf, o[t]);
    }
  }

  float inv[8];
#pragma unroll
  for (int g = 0; g < 8; ++g) inv[g] = 1.0f / l[g];
  float* ob = O + ((size_t)bh * SEQ + q0) * DH2;
#pragma unroll
  for (int t = 0; t < 16; ++t)
#pragma unroll
    for (int g = 0; g < 8; ++g){
      int row = (lane >> 4) * 8 + g;
      int col = t*16 + (lane & 15);
      ob[(size_t)row * DH2 + col] = o[t][g] * inv[g];
    }
}

// ----------------------- diff + RMS-norm + scale, emit bf16 (B,N,2048) layout
__global__ __launch_bounds__(128) void combine_kernel(const float* __restrict__ O1,
                                                      const float* __restrict__ O2,
                                                      const float* __restrict__ lam,
                                                      const float* __restrict__ rms_scale,
                                                      us* __restrict__ Ob){
  int lane = threadIdx.x & 31;
  int w  = blockIdx.x * 4 + (threadIdx.x >> 5);   // row index = bh*SEQ + n
  int n  = w & (SEQ - 1);
  int bh = w >> 11;
  int h  = bh & (HEADS - 1);
  int b  = bh >> 3;
  const float* p1 = O1 + (size_t)w * DH2;
  const float* p2 = O2 + (size_t)w * DH2;
  float lm = lam[h];
  float v[8], ss = 0.f;
#pragma unroll
  for (int j = 0; j < 8; ++j){
    int c = lane + j*32;
    float x = p1[c] - lm * p2[c];
    v[j] = x;
    ss += x * x;
  }
#pragma unroll
  for (int off = 1; off < 32; off <<= 1) ss += __shfl_xor(ss, off, 32);
  float scale = (1.0f - LAMBDA_INIT) / sqrtf(ss * (1.0f / DH2) + RMS_EPS);
  us* dst = Ob + (size_t)(b*SEQ + n) * DE + h * DH2;
#pragma unroll
  for (int j = 0; j < 8; ++j){
    int c = lane + j*32;
    dst[c] = f2bf(v[j] * scale * rms_scale[c]);
  }
}

// --------------------------------------------------------------------- driver
extern "C" void kernel_launch(void* const* d_in, const int* in_sizes, int n_in,
                              void* d_out, int out_size, void* d_ws, size_t ws_size,
                              hipStream_t stream){
  (void)in_sizes; (void)n_in; (void)out_size; (void)ws_size;
  const float* X   = (const float*)d_in[0];
  const float* Wq  = (const float*)d_in[1];
  const float* Wk  = (const float*)d_in[2];
  const float* Wv  = (const float*)d_in[3];
  const float* Wo  = (const float*)d_in[4];
  const float* lq1 = (const float*)d_in[5];
  const float* lk1 = (const float*)d_in[6];
  const float* lq2 = (const float*)d_in[7];
  const float* lk2 = (const float*)d_in[8];
  const float* rs  = (const float*)d_in[9];
  float* out = (float*)d_out;

  char* ws = (char*)d_ws;
  size_t off = 0;
  auto alloc = [&](size_t bytes)->char*{
    char* p = ws + off; off += (bytes + 255) & ~(size_t)255; return p;
  };
  float* lam = (float*)alloc(HEADS * sizeof(float));
  us* Xb  = (us*)alloc((size_t)BATCH*SEQ*DMODEL*2);
  us* Wqb = (us*)alloc((size_t)DE*DMODEL*2);
  us* Wkb = (us*)alloc((size_t)DE*DMODEL*2);
  us* Wvb = (us*)alloc((size_t)DE*DMODEL*2);
  us* Wob = (us*)alloc((size_t)DMODEL*DE*2);
  us* Qb  = (us*)alloc((size_t)BATCH*HEADS*SEQ*DH2*2);
  us* Kb  = (us*)alloc((size_t)BATCH*HEADS*SEQ*DH2*2);
  us* Vb  = (us*)alloc((size_t)BATCH*HEADS*SEQ*DH2*2);
  float* O1 = (float*)alloc((size_t)BATCH*HEADS*SEQ*DH2*4);
  float* O2 = (float*)alloc((size_t)BATCH*HEADS*SEQ*DH2*4);
  us* Ob  = (us*)alloc((size_t)BATCH*SEQ*DE*2);

  auto cvt = [&](const float* src, us* dst, int n){
    cvt_bf16<<<dim3((n + 255)/256), dim3(256), 0, stream>>>(src, dst, n);
  };
  cvt(X,  Xb,  BATCH*SEQ*DMODEL);
  cvt(Wq, Wqb, DE*DMODEL);
  cvt(Wk, Wkb, DE*DMODEL);
  cvt(Wv, Wvb, DE*DMODEL);
  cvt(Wo, Wob, DMODEL*DE);
  lambda_kernel<<<dim3(1), dim3(32), 0, stream>>>(lq1, lk1, lq2, lk2, lam);

  dim3 g1(BATCH*SEQ/128, DE/128);
  gemm_bf16_tn<<<g1, dim3(256), 0, stream>>>(Xb, Wqb, DMODEL, DE, 0, Qb, nullptr);
  gemm_bf16_tn<<<g1, dim3(256), 0, stream>>>(Xb, Wkb, DMODEL, DE, 0, Kb, nullptr);
  gemm_bf16_tn<<<g1, dim3(256), 0, stream>>>(Xb, Wvb, DMODEL, DE, 0, Vb, nullptr);

  dim3 gfa(SEQ/64, BATCH*HEADS);
  diff_attn_pass<<<gfa, dim3(128), 0, stream>>>(Qb, Kb, Vb, O1, 0);
  diff_attn_pass<<<gfa, dim3(128), 0, stream>>>(Qb, Kb, Vb, O2, 1);

  combine_kernel<<<dim3(BATCH*HEADS*SEQ/4), dim3(128), 0, stream>>>(O1, O2, lam, rs, Ob);

  dim3 g2(BATCH*SEQ/128, DMODEL/128);
  gemm_bf16_tn<<<g2, dim3(256), 0, stream>>>(Ob, Wob, DE, DMODEL, 1, nullptr, out);
}
